// AgnisV5_25907242729991
// MI455X (gfx1250) — matrix operand order repaired and checked by reading, not compiled, not run on hardware
//
#include <hip/hip_runtime.h>
#include <hip/hip_bf16.h>
#include <math.h>
#include <stdint.h>

#define B_ROWS 4096
#define DIM    768
#define VOCAB  50257

typedef __bf16 bf16_t;
typedef __attribute__((ext_vector_type(16))) __bf16 v16bf;
typedef __attribute__((ext_vector_type(8)))  __bf16 v8bf;
typedef __attribute__((ext_vector_type(8)))  float  v8f;

// ---------------------------------------------------------------------------
// WMMA fragment loaders (wave32, contiguous k order shared by A and B, 16-bit A 16x32 / B 32x16).
// A: lane L<16 holds row M=L, K k0..k0+15; lane L>=16 holds
//    row M=L-16, K k0+16..k0+31.  Two 16B vector loads.
// B (NT: W rows are [N][K]): lane L<16 holds col N=L, K k0..k0+15;
//    lane L>=16 holds col N=L-16, K k0+16..k0+31.  Two 16B vector loads.
// ---------------------------------------------------------------------------
__device__ inline v16bf cat8(v8bf lo, v8bf hi) {
  v16bf r;
#pragma unroll
  for (int i = 0; i < 8; ++i) { r[i] = lo[i]; r[i + 8] = hi[i]; }
  return r;
}

__device__ inline v16bf zero16bf() {
  v16bf r;
#pragma unroll
  for (int i = 0; i < 16; ++i) r[i] = (bf16_t)0.0f;
  return r;
}

__device__ inline v16bf load_a_frag(const bf16_t* A, int lda, int m0, int k0, int lane) {
  const bf16_t* p = A + (size_t)(m0 + (lane & 15)) * lda + k0 + (lane >> 4) * 16;
  return cat8(*(const v8bf*)p, *(const v8bf*)(p + 8));
}

__device__ inline v16bf load_b_full(const bf16_t* W, int ldw, int n0, int k0, int lane) {
  const bf16_t* p = W + (size_t)(n0 + (lane & 15)) * ldw + k0 + (lane >> 4) * 16;
  return cat8(*(const v8bf*)p, *(const v8bf*)(p + 8));
}

__device__ inline v16bf load_b_guard(const bf16_t* W, int ldw, int n0, int k0, int lane, int N) {
  int nc = n0 + (lane & 15);
  if (nc < N) {
    const bf16_t* p = W + (size_t)nc * ldw + k0 + (lane >> 4) * 16;
    return cat8(*(const v8bf*)p, *(const v8bf*)(p + 8));
  }
  return zero16bf();
}

// ---------------------------------------------------------------------------
// Generic WMMA GEMM:  C[M,N] = epilogue( A[M,K] * Wrows[N,K]^T + bias )
// block = 128 threads (4 waves); wave w -> rows [by*128+32w, +32), cols
// [bx*64, +64).  Ping-pong double buffered over K (K % 64 == 0, K >= 64).
// modes: 0 plain->f32 | 1 GELU->bf16 | 2 plain->f32+bf16 | 3 sigmoid->f32
//        4 eps = aux - acc, stored transposed bf16 | 5 plain->bf16
//        6 plain->f32 non-temporal store (logits: keep L2 for embedding)
// ---------------------------------------------------------------------------
struct GemmArgs {
  const bf16_t* A;
  const bf16_t* W;
  int M, N, K;
  const float* bias;   // nullable
  int mode;
  float* outF;         // f32 output, ld = N
  bf16_t* outB;        // bf16 output
  int ldB;
  int colOff;
  const float* aux;    // mode 4: coreN_f32, ld = N
  bf16_t* outT;        // mode 4: transposed bf16 out
  int ldT;
};

struct Frags { v16bf a0, a1, b0, b1, b2, b3; };

template <bool FULL>
__device__ inline Frags load_frags(const GemmArgs& g, int m0, int n0, int k0, int lane) {
  Frags f;
  f.a0 = load_a_frag(g.A, g.K, m0,      k0, lane);
  f.a1 = load_a_frag(g.A, g.K, m0 + 16, k0, lane);
  if (FULL) {
    f.b0 = load_b_full(g.W, g.K, n0,      k0, lane);
    f.b1 = load_b_full(g.W, g.K, n0 + 16, k0, lane);
    f.b2 = load_b_full(g.W, g.K, n0 + 32, k0, lane);
    f.b3 = load_b_full(g.W, g.K, n0 + 48, k0, lane);
  } else {
    f.b0 = load_b_guard(g.W, g.K, n0,      k0, lane, g.N);
    f.b1 = load_b_guard(g.W, g.K, n0 + 16, k0, lane, g.N);
    f.b2 = load_b_guard(g.W, g.K, n0 + 32, k0, lane, g.N);
    f.b3 = load_b_guard(g.W, g.K, n0 + 48, k0, lane, g.N);
  }
  return f;
}

__device__ inline void mma8(const Frags& f, v8f acc0[4], v8f acc1[4]) {
  acc0[0] = __builtin_amdgcn_wmma_f32_16x16x32_bf16(false, f.a0, false, f.b0, (short)0, acc0[0], false, false);
  acc1[0] = __builtin_amdgcn_wmma_f32_16x16x32_bf16(false, f.a1, false, f.b0, (short)0, acc1[0], false, false);
  acc0[1] = __builtin_amdgcn_wmma_f32_16x16x32_bf16(false, f.a0, false, f.b1, (short)0, acc0[1], false, false);
  acc1[1] = __builtin_amdgcn_wmma_f32_16x16x32_bf16(false, f.a1, false, f.b1, (short)0, acc1[1], false, false);
  acc0[2] = __builtin_amdgcn_wmma_f32_16x16x32_bf16(false, f.a0, false, f.b2, (short)0, acc0[2], false, false);
  acc1[2] = __builtin_amdgcn_wmma_f32_16x16x32_bf16(false, f.a1, false, f.b2, (short)0, acc1[2], false, false);
  acc0[3] = __builtin_amdgcn_wmma_f32_16x16x32_bf16(false, f.a0, false, f.b3, (short)0, acc0[3], false, false);
  acc1[3] = __builtin_amdgcn_wmma_f32_16x16x32_bf16(false, f.a1, false, f.b3, (short)0, acc1[3], false, false);
}

template <bool FULL>
__device__ inline void gemm_loop(const GemmArgs& g, int m0, int n0, int lane,
                                 v8f acc0[4], v8f acc1[4]) {
  // K % 64 == 0 and K >= 64 for every GEMM in this pipeline.
  Frags cur = load_frags<FULL>(g, m0, n0, 0, lane);
  int k0 = 0;
  for (; k0 < g.K - 64; k0 += 64) {
    Frags nxt = load_frags<FULL>(g, m0, n0, k0 + 32, lane);
    mma8(cur, acc0, acc1);
    cur = load_frags<FULL>(g, m0, n0, k0 + 64, lane);
    mma8(nxt, acc0, acc1);
  }
  Frags nxt = load_frags<FULL>(g, m0, n0, k0 + 32, lane);
  mma8(cur, acc0, acc1);
  mma8(nxt, acc0, acc1);
}

__device__ inline void epilogue_16x64(const GemmArgs& g, int mbase, int n0, int lane,
                                      v8f acc[4]) {
  const int hl = lane >> 4;
  const int nl = lane & 15;
#pragma unroll
  for (int j = 0; j < 4; ++j) {
    const int col = n0 + 16 * j + nl;
    if (col >= g.N) continue;
    const float bc = g.bias ? g.bias[col] : 0.0f;
#pragma unroll
    for (int r = 0; r < 8; ++r) {
      const int row = mbase + 8 * hl + r;
      float v = acc[j][r] + bc;
      if (g.mode == 6) {
        // streaming store: don't let the 823MB logits evict the L2-resident
        // bf16 embedding table
        __builtin_nontemporal_store(v, &g.outF[(size_t)row * g.N + col]);
      } else if (g.mode == 0) {
        g.outF[(size_t)row * g.N + col] = v;
      } else if (g.mode == 1) {
        float ge = 0.5f * v * (1.0f + erff(v * 0.70710678118654752f));
        g.outB[(size_t)row * g.ldB + g.colOff + col] = (bf16_t)ge;
      } else if (g.mode == 2) {
        g.outF[(size_t)row * g.N + col] = v;
        g.outB[(size_t)row * g.ldB + g.colOff + col] = (bf16_t)v;
      } else if (g.mode == 3) {
        g.outF[(size_t)row * g.N + col] = 1.0f / (1.0f + expf(-v));
      } else if (g.mode == 4) {
        float e = g.aux[(size_t)row * g.N + col] - v;
        g.outT[(size_t)col * g.ldT + row] = (bf16_t)e;
      } else {  // 5
        g.outB[(size_t)row * g.ldB + g.colOff + col] = (bf16_t)v;
      }
    }
  }
}

__global__ __launch_bounds__(128) void wmma_gemm_kernel(GemmArgs g) {
  const int lane = threadIdx.x & 31;
  const int wv   = threadIdx.x >> 5;
  const int m0 = blockIdx.y * 128 + wv * 32;  // M is always a multiple of 128
  const int n0 = blockIdx.x * 64;

  v8f acc0[4], acc1[4];
#pragma unroll
  for (int j = 0; j < 4; ++j)
#pragma unroll
    for (int r = 0; r < 8; ++r) { acc0[j][r] = 0.0f; acc1[j][r] = 0.0f; }

  if (n0 + 64 <= g.N) {
    gemm_loop<true>(g, m0, n0, lane, acc0, acc1);
  } else {
    gemm_loop<false>(g, m0, n0, lane, acc0, acc1);
  }

  epilogue_16x64(g, m0,      n0, lane, acc0);
  epilogue_16x64(g, m0 + 16, n0, lane, acc1);
}

// ---------------------------------------------------------------------------
// Helpers
// ---------------------------------------------------------------------------
__device__ inline float block_reduce_sum256(float v, float* sred) {
  int tid = threadIdx.x;
  sred[tid] = v;
  __syncthreads();
  for (int s = 128; s > 0; s >>= 1) {
    if (tid < s) sred[tid] += sred[tid + s];
    __syncthreads();
  }
  float r = sred[0];
  __syncthreads();
  return r;
}

// Row prep: gather+L2norm(emb) -> embN_f32 + gate_in[:, :768] bf16,
//           L2norm(core_raw) -> coreN_f32 + coreN_bf16, h_prev -> bf16.
__global__ __launch_bounds__(256) void prep_kernel(
    const int* tok, const float* emb, const float* core_raw, const float* h_prev,
    float* embN, bf16_t* gate_in, float* coreN_f, bf16_t* coreN_b, bf16_t* hprev_b) {
  __shared__ float sred[256];
  const int row = blockIdx.x;
  const int tid = threadIdx.x;
  const float* er = emb + (size_t)tok[row] * DIM;

  float ev[3];
  float ss = 0.0f;
#pragma unroll
  for (int j = 0; j < 3; ++j) { ev[j] = er[tid + 256 * j]; ss += ev[j] * ev[j]; }
  float tot = block_reduce_sum256(ss, sred);
  float sc = 1.0f / fmaxf(sqrtf(tot), 1e-12f);
#pragma unroll
  for (int j = 0; j < 3; ++j) {
    int d = tid + 256 * j;
    float v = ev[j] * sc;
    embN[(size_t)row * DIM + d] = v;
    gate_in[(size_t)row * (2 * DIM) + d] = (bf16_t)v;
  }

  float cv[3];
  ss = 0.0f;
#pragma unroll
  for (int j = 0; j < 3; ++j) {
    cv[j] = core_raw[(size_t)row * DIM + tid + 256 * j];
    ss += cv[j] * cv[j];
  }
  tot = block_reduce_sum256(ss, sred);
  sc = 1.0f / fmaxf(sqrtf(tot), 1e-12f);
#pragma unroll
  for (int j = 0; j < 3; ++j) {
    int d = tid + 256 * j;
    float v = cv[j] * sc;
    coreN_f[(size_t)row * DIM + d] = v;
    coreN_b[(size_t)row * DIM + d] = (bf16_t)v;
    hprev_b[(size_t)row * DIM + d] = (bf16_t)h_prev[(size_t)row * DIM + d];
  }
}

// LDS-tiled transpose f32 -> bf16; rows, cols multiples of 32; block (32,8)
__global__ __launch_bounds__(256) void transpose_f2b_kernel(
    const float* in, bf16_t* out, int rows, int cols) {
  __shared__ float tile[32][33];
  const int c0 = blockIdx.x * 32, r0 = blockIdx.y * 32;
  for (int i = threadIdx.y; i < 32; i += 8)
    tile[i][threadIdx.x] = in[(size_t)(r0 + i) * cols + (c0 + threadIdx.x)];
  __syncthreads();
  for (int i = threadIdx.y; i < 32; i += 8)
    out[(size_t)(c0 + i) * rows + (r0 + threadIdx.x)] = (bf16_t)tile[threadIdx.x][i];
}

// grid-stride f32 -> bf16 convert; non-temporal reads (source never reused)
__global__ __launch_bounds__(256) void f2b_kernel(const float* in, bf16_t* out, size_t n) {
  size_t i = blockIdx.x * (size_t)blockDim.x + threadIdx.x;
  size_t stride = (size_t)gridDim.x * blockDim.x;
  for (; i < n; i += stride)
    out[i] = (bf16_t)__builtin_nontemporal_load(&in[i]);
}

// R_newT[j][k] = bf16(clip(0.999*R[k][j] + (eta*surprise/B)*dR_sum[k][j], +-3))
__global__ __launch_bounds__(256) void rnew_kernel(const float* R, const float* dR, bf16_t* RnewT) {
  size_t n = (size_t)DIM * DIM;
  size_t i = blockIdx.x * (size_t)blockDim.x + threadIdx.x;
  size_t stride = (size_t)gridDim.x * blockDim.x;
  for (; i < n; i += stride) {
    int k = (int)(i / DIM);
    int j = (int)(i - (size_t)k * DIM);
    float v = 0.999f * R[i] + (0.005f / 4096.0f) * dR[i];
    v = fminf(3.0f, fmaxf(-3.0f, v));
    RnewT[(size_t)j * DIM + k] = (bf16_t)v;
  }
}

// h_t = g*(core_feat + 0.4*temporal) + (1-g)*emb ; LayerNorm -> fused bf16
__global__ __launch_bounds__(256) void fuse_ln_kernel(
    const float* gate, const float* corefeat, const float* temporal,
    const float* embN, const float* gamma, const float* beta, bf16_t* fused) {
  __shared__ float sred[256];
  const int row = blockIdx.x;
  const int tid = threadIdx.x;
  float hv[3];
  float s = 0.0f;
#pragma unroll
  for (int j = 0; j < 3; ++j) {
    size_t idx = (size_t)row * DIM + tid + 256 * j;
    float g = gate[idx];
    float h = g * (corefeat[idx] + 0.4f * temporal[idx]) + (1.0f - g) * embN[idx];
    hv[j] = h;
    s += h;
  }
  float mu = block_reduce_sum256(s, sred) * (1.0f / (float)DIM);
  s = 0.0f;
#pragma unroll
  for (int j = 0; j < 3; ++j) { float d = hv[j] - mu; s += d * d; }
  float var = block_reduce_sum256(s, sred) * (1.0f / (float)DIM);
  float rstd = rsqrtf(var + 1e-5f);
#pragma unroll
  for (int j = 0; j < 3; ++j) {
    int d = tid + 256 * j;
    fused[(size_t)row * DIM + d] = (bf16_t)((hv[j] - mu) * rstd * gamma[d] + beta[d]);
  }
}

// ---------------------------------------------------------------------------
// Host
// ---------------------------------------------------------------------------
static inline void launch_gemm(const GemmArgs& g, hipStream_t stream) {
  dim3 grid((g.N + 63) / 64, g.M / 128);
  wmma_gemm_kernel<<<grid, 128, 0, stream>>>(g);
}

extern "C" void kernel_launch(void* const* d_in, const int* in_sizes, int n_in,
                              void* d_out, int out_size, void* d_ws, size_t ws_size,
                              hipStream_t stream) {
  (void)in_sizes; (void)n_in; (void)out_size; (void)ws_size;
  const int*   tok      = (const int*)d_in[0];
  const float* core_raw = (const float*)d_in[1];
  const float* h_prev   = (const float*)d_in[2];
  const float* embedding= (const float*)d_in[3];
  const float* W1       = (const float*)d_in[4];
  const float* b1       = (const float*)d_in[5];
  const float* W2       = (const float*)d_in[6];
  const float* b2       = (const float*)d_in[7];
  const float* Wg       = (const float*)d_in[8];
  const float* bg       = (const float*)d_in[9];
  const float* Wt       = (const float*)d_in[10];
  const float* R        = (const float*)d_in[11];
  const float* gamma    = (const float*)d_in[12];
  const float* beta     = (const float*)d_in[13];
  float* logits = (float*)d_out;

  const size_t BD = (size_t)B_ROWS * DIM;
  const size_t DD = (size_t)DIM * DIM;
  const size_t VD = (size_t)VOCAB * DIM;

  char* w = (char*)d_ws;
  auto alloc = [&](size_t bytes) {
    char* p = w;
    w += (bytes + 255) & ~(size_t)255;
    return p;
  };
  bf16_t* gate_in   = (bf16_t*)alloc(BD * 2 * sizeof(bf16_t)); // emb | core_feat (bf16)
  float*  embN_f32  = (float*) alloc(BD * sizeof(float));
  float*  coreN_f32 = (float*) alloc(BD * sizeof(float));
  bf16_t* coreN_bf  = (bf16_t*)alloc(BD * sizeof(bf16_t));
  bf16_t* hprev_bf  = (bf16_t*)alloc(BD * sizeof(bf16_t));
  bf16_t* hprevT_bf = (bf16_t*)alloc(BD * sizeof(bf16_t));   // [DIM, B]
  bf16_t* gelu1_bf  = (bf16_t*)alloc(BD * sizeof(bf16_t));
  float*  cfeat_f32 = (float*) alloc(BD * sizeof(float));
  bf16_t* Rt_bf     = (bf16_t*)alloc(DD * sizeof(bf16_t));   // R^T
  bf16_t* epsT_bf   = (bf16_t*)alloc(BD * sizeof(bf16_t));   // [DIM, B]
  float*  dR_f32    = (float*) alloc(DD * sizeof(float));    // raw sum (no /B)
  bf16_t* RnewT_bf  = (bf16_t*)alloc(DD * sizeof(bf16_t));
  bf16_t* t1_bf     = (bf16_t*)alloc(BD * sizeof(bf16_t));
  float*  temp_f32  = (float*) alloc(BD * sizeof(float));
  float*  gate_f32  = (float*) alloc(BD * sizeof(float));
  bf16_t* fused_bf  = (bf16_t*)alloc(BD * sizeof(bf16_t));
  bf16_t* W1_bf     = (bf16_t*)alloc(DD * sizeof(bf16_t));
  bf16_t* W2_bf     = (bf16_t*)alloc(DD * sizeof(bf16_t));
  bf16_t* Wt_bf     = (bf16_t*)alloc(DD * sizeof(bf16_t));
  bf16_t* Wg_bf     = (bf16_t*)alloc(DD * 2 * sizeof(bf16_t));
  bf16_t* emb_bf    = (bf16_t*)alloc(VD * sizeof(bf16_t));   // fits in 192MB L2

  // --- bf16 conversions (weights + embedding table) ---
  f2b_kernel<<<1024, 256, 0, stream>>>(W1, W1_bf, DD);
  f2b_kernel<<<1024, 256, 0, stream>>>(W2, W2_bf, DD);
  f2b_kernel<<<1024, 256, 0, stream>>>(Wt, Wt_bf, DD);
  f2b_kernel<<<2048, 256, 0, stream>>>(Wg, Wg_bf, DD * 2);
  f2b_kernel<<<8192, 256, 0, stream>>>(embedding, emb_bf, VD);

  // --- transposes (for the NN-layout products) ---
  transpose_f2b_kernel<<<dim3(DIM / 32, DIM / 32), dim3(32, 8), 0, stream>>>(R, Rt_bf, DIM, DIM);
  transpose_f2b_kernel<<<dim3(DIM / 32, B_ROWS / 32), dim3(32, 8), 0, stream>>>(h_prev, hprevT_bf, B_ROWS, DIM);

  // --- row prep: gather + L2 normalize + bf16 packs ---
  prep_kernel<<<B_ROWS, 256, 0, stream>>>(tok, embedding, core_raw, h_prev,
                                          embN_f32, gate_in, coreN_f32, coreN_bf, hprev_bf);

  GemmArgs g{};
  // G1: gelu1 = GELU(coreN @ W1^T + b1) -> bf16
  g = {coreN_bf, W1_bf, B_ROWS, DIM, DIM, b1, 1, nullptr, gelu1_bf, DIM, 0, nullptr, nullptr, 0};
  launch_gemm(g, stream);
  // G2: core_feat = gelu1 @ W2^T + b2 -> f32 + bf16 into gate_in[:, 768:]
  g = {gelu1_bf, W2_bf, B_ROWS, DIM, DIM, b2, 2, cfeat_f32, gate_in, 2 * DIM, DIM, nullptr, nullptr, 0};
  launch_gemm(g, stream);
  // G3: x_hat = h_prev @ R ; epsT[j][b] = coreN[b][j] - x_hat[b][j]
  g = {hprev_bf, Rt_bf, B_ROWS, DIM, DIM, nullptr, 4, nullptr, nullptr, 0, 0, coreN_f32, epsT_bf, B_ROWS};
  launch_gemm(g, stream);
  // G4: dR_sum = h_prev^T @ eps  (M=N=768, K=4096)
  g = {hprevT_bf, epsT_bf, DIM, DIM, B_ROWS, nullptr, 0, dR_f32, nullptr, 0, 0, nullptr, nullptr, 0};
  launch_gemm(g, stream);
  // R_newT = clip(0.999 R + eta/B * dR)^T -> bf16
  rnew_kernel<<<1024, 256, 0, stream>>>(R, dR_f32, RnewT_bf);
  // G5: t1 = h_prev @ R_new -> bf16
  g = {hprev_bf, RnewT_bf, B_ROWS, DIM, DIM, nullptr, 5, nullptr, t1_bf, DIM, 0, nullptr, nullptr, 0};
  launch_gemm(g, stream);
  // G6: temporal = t1 @ Wt^T -> f32
  g = {t1_bf, Wt_bf, B_ROWS, DIM, DIM, nullptr, 0, temp_f32, nullptr, 0, 0, nullptr, nullptr, 0};
  launch_gemm(g, stream);
  // G7: gate = sigmoid([emb, core_feat] @ Wg^T + bg)  (K=1536)
  g = {gate_in, Wg_bf, B_ROWS, DIM, 2 * DIM, bg, 3, gate_f32, nullptr, 0, 0, nullptr, nullptr, 0};
  launch_gemm(g, stream);

  // blend + LayerNorm -> fused bf16
  fuse_ln_kernel<<<B_ROWS, 256, 0, stream>>>(gate_f32, cfeat_f32, temp_f32,
                                             embN_f32, gamma, beta, fused_bf);

  // G8: logits = fused @ embedding^T (N=50257, tail guard, NT streaming stores)
  g = {fused_bf, emb_bf, B_ROWS, VOCAB, DIM, nullptr, 6, logits, nullptr, 0, 0, nullptr, nullptr, 0};
  launch_gemm(g, stream);
}
